// Contrastive_22900765622580
// MI455X (gfx1250) — matrix operand order repaired and checked by reading, not compiled, not run on hardware
//
#include <hip/hip_runtime.h>
#include <hip/hip_bf16.h>
#include <stdint.h>

#define ALPHA_C 1.0f

typedef __bf16        v16bf __attribute__((ext_vector_type(16)));
typedef float         v8f   __attribute__((ext_vector_type(8)));
typedef unsigned int  u32x4 __attribute__((ext_vector_type(4)));
typedef int           i32x4 __attribute__((ext_vector_type(4)));
typedef int           i32x8 __attribute__((ext_vector_type(8)));

struct Packed32 { uint4 a, b; };   // 32 bytes == v16bf

__device__ __forceinline__ v8f wmma_bf16(v16bf a, v16bf b, v8f c) {
    // (neg_a, A, neg_b, B, c_mod, C, reuse_a, reuse_b)
    return __builtin_amdgcn_wmma_f32_16x16x32_bf16(false, a, false, b,
                                                   (short)0, c, false, false);
}

// Load a 16-element bf16 fragment from LDS as two 16B chunks (ds_load_b128 x2).
__device__ __forceinline__ v16bf load_frag(const unsigned short* p, int off0, int off1) {
    Packed32 q;
    q.a = *(const uint4*)(p + off0);
    q.b = *(const uint4*)(p + off1);
    return __builtin_bit_cast(v16bf, q);
}

// Split fp32 into bf16 hi (RNE) + bf16 lo (RNE of residual). Pure integer path.
__device__ __forceinline__ void bf16_split(float f, unsigned int& h, unsigned int& l) {
    unsigned int u  = __builtin_bit_cast(unsigned int, f);
    unsigned int r  = u + 0x7FFFu + ((u >> 16) & 1u);
    unsigned int hh = r >> 16;
    float hf = __builtin_bit_cast(float, hh << 16);
    float d  = f - hf;
    unsigned int u2 = __builtin_bit_cast(unsigned int, d);
    unsigned int r2 = u2 + 0x7FFFu + ((u2 >> 16) & 1u);
    h = hh;
    l = r2 >> 16;
}

#define LDS_STRIDE 72     // bf16 tile row stride in halfwords (144B, bank-spread)
#define STG_STRIDE 68     // fp32 staging row stride in dwords (TDM pad: 64 data + 4 pad)

// Issue one TDM 2-D tile load: 64 rows x 64 fp32 from gptr (row stride K floats)
// into LDS at byte offset lds_off, padded to STG_STRIDE dwords per row.
__device__ __forceinline__ void tdm_stage_tile(unsigned lds_off, const float* gptr, int K) {
    unsigned long long ga = (unsigned long long)(uintptr_t)gptr;
    u32x4 g0;
    g0[0] = 1u;                                           // count=1, user descriptor
    g0[1] = lds_off;                                      // lds_addr (bytes)
    g0[2] = (unsigned)(ga & 0xFFFFFFFFu);                 // global_addr[31:0]
    g0[3] = (unsigned)((ga >> 32) & 0x01FFFFFFu)          // global_addr[56:32]
          | (2u << 30);                                   // type = 2 ("image")
    i32x8 g1;
    g1[0] = (int)((2u << 16)                              // data_size = 4 bytes
                | (1u << 20)                              // pad_enable
                | (5u << 22)                              // pad_interval: every 64 dwords
                | (3u << 25));                            // pad_amount: 4 dwords
    g1[1] = (int)(((unsigned)K & 0xFFFFu) << 16);         // tensor_dim0[15:0] (atomic addr = 0)
    g1[2] = (int)(((unsigned)K >> 16) | (64u << 16));     // tensor_dim0[31:16] | tensor_dim1[15:0]=64
    g1[3] = (int)(64u << 16);                             // tensor_dim1 hi = 0 | tile_dim0 = 64
    g1[4] = 64;                                           // tile_dim1 = 64 (tile_dim2 = 0 -> 2D)
    g1[5] = K;                                            // tensor_dim0_stride[31:0]
    g1[6] = 0;                                            // stride hi / dim1_stride lo
    g1[7] = 0;
    i32x4 z4 = {0, 0, 0, 0};
#if __clang_major__ >= 23
    i32x8 z8 = {0, 0, 0, 0, 0, 0, 0, 0};
    __builtin_amdgcn_tensor_load_to_lds(g0, g1, z4, z4, z8, 0);
#else
    __builtin_amdgcn_tensor_load_to_lds(g0, g1, z4, z4, 0);
#endif
}

// out[64x64] = scale * X[64xK] * X[64xK]^T per block. K % 64 == 0.
// TDM double-buffers raw fp32 tiles into LDS; threads split fp32 -> bf16 hi/lo
// tiles in LDS; 8 wave32s run 12 v_wmma_f32_16x16x32_bf16 per 64-wide K chunk.
__global__ __launch_bounds__(256)
void gram64_bf16x2_kernel(const float* __restrict__ X, float* __restrict__ out,
                          int K, float scale) {
    __shared__ __align__(16) float          stage[2][64 * STG_STRIDE];
    __shared__ __align__(16) unsigned short sh_hi[64 * LDS_STRIDE];
    __shared__ __align__(16) unsigned short sh_lo[64 * LDS_STRIDE];

    const float* Xb = X + (size_t)blockIdx.x * 64u * (size_t)K;
    float*       Ob = out + (size_t)blockIdx.x * 4096u;

    const int tid  = threadIdx.x;
    // ---- convert-phase mapping: 4 threads per row, 16 consecutive fp32 each ----
    const int lrow = tid >> 2;            // 0..63
    const int lcol = (tid & 3) << 4;      // 0,16,32,48

    // ---- WMMA mapping: 8 waves cover the 4x4 grid of 16x16 tiles ----
    const int lane  = tid & 31;
    const int wv    = tid >> 5;
    const int rb    = wv & 3;             // A row-block (M tile)
    const int cp    = wv >> 2;            // column-pair: tiles {2cp, 2cp+1}
    const int l15   = lane & 15;
    const int aRow  = 16 * rb + l15;
    const int aHalf = (lane < 16) ? 0 : 16;   // A layout: hi lanes hold K 16..31
    const int bRow0 = 32 * cp + l15;          // B col j -> row j of X (A*A^T)
    const int bRow1 = bRow0 + 16;
    const int bHalf = (lane < 16) ? 0 : 16;   // B layout: hi lanes hold K 16..31

    const unsigned stageOff[2] = {
        (unsigned)(uintptr_t)&stage[0][0],    // generic addr low 32 bits == LDS byte offset
        (unsigned)(uintptr_t)&stage[1][0]
    };

    v8f acc0 = {};
    v8f acc1 = {};

    const int nChunks = K >> 6;

    // Prologue: stage chunk 0.
    if (wv == 0) tdm_stage_tile(stageOff[0], Xb, K);

    for (int c = 0; c < nChunks; ++c) {
        const int buf = c & 1;

        // Overlap: kick chunk c+1 into the other buffer, then ensure chunk c landed.
        if (wv == 0) {
            if (c + 1 < nChunks) {
                tdm_stage_tile(stageOff[buf ^ 1], Xb + (size_t)(c + 1) * 64u, K);
                __builtin_amdgcn_s_wait_tensorcnt(1);   // TDM in-order: chunk c complete
            } else {
                __builtin_amdgcn_s_wait_tensorcnt(0);
            }
        }
        __syncthreads();   // publish staged chunk c; prior bf16-tile reads also done

        // -------- fp32 (staged in LDS) -> (bf16_hi, bf16_lo) tiles --------
        {
            const float* sp = &stage[buf][lrow * STG_STRIDE + lcol];
            float4 f0 = *(const float4*)(sp + 0);
            float4 f1 = *(const float4*)(sp + 4);
            float4 f2 = *(const float4*)(sp + 8);
            float4 f3 = *(const float4*)(sp + 12);

            unsigned int hw[8], lw[8];
            float e[16] = { f0.x, f0.y, f0.z, f0.w, f1.x, f1.y, f1.z, f1.w,
                            f2.x, f2.y, f2.z, f2.w, f3.x, f3.y, f3.z, f3.w };
#pragma unroll
            for (int q = 0; q < 8; ++q) {
                unsigned int h0, l0, h1, l1;
                bf16_split(e[2 * q + 0], h0, l0);
                bf16_split(e[2 * q + 1], h1, l1);
                hw[q] = h0 | (h1 << 16);
                lw[q] = l0 | (l1 << 16);
            }
            unsigned short* dh = &sh_hi[lrow * LDS_STRIDE + lcol];
            unsigned short* dl = &sh_lo[lrow * LDS_STRIDE + lcol];
            *(uint4*)(dh + 0) = make_uint4(hw[0], hw[1], hw[2], hw[3]);
            *(uint4*)(dh + 8) = make_uint4(hw[4], hw[5], hw[6], hw[7]);
            *(uint4*)(dl + 0) = make_uint4(lw[0], lw[1], lw[2], lw[3]);
            *(uint4*)(dl + 8) = make_uint4(lw[4], lw[5], lw[6], lw[7]);
        }
        __syncthreads();   // bf16 tiles ready

        // -------- WMMA over the two 32-wide k-steps of this chunk --------
#pragma unroll
        for (int s = 0; s < 64; s += 32) {
            const int ao = aRow * LDS_STRIDE + s + aHalf;
            v16bf Ah = load_frag(sh_hi, ao, ao + 8);
            v16bf Al = load_frag(sh_lo, ao, ao + 8);

            const int b0 = bRow0 * LDS_STRIDE + s + bHalf;
            const int b1 = bRow1 * LDS_STRIDE + s + bHalf;
            v16bf Bh0 = load_frag(sh_hi, b0, b0 + 8);
            v16bf Bl0 = load_frag(sh_lo, b0, b0 + 8);
            v16bf Bh1 = load_frag(sh_hi, b1, b1 + 8);
            v16bf Bl1 = load_frag(sh_lo, b1, b1 + 8);

            // fp32 ~= hi*hi + hi*lo + lo*hi  (lo*lo below rounding)
            acc0 = wmma_bf16(Ah, Bh0, acc0);
            acc0 = wmma_bf16(Ah, Bl0, acc0);
            acc0 = wmma_bf16(Al, Bh0, acc0);
            acc1 = wmma_bf16(Ah, Bh1, acc1);
            acc1 = wmma_bf16(Ah, Bl1, acc1);
            acc1 = wmma_bf16(Al, Bh1, acc1);
        }
    }

    // -------- epilogue: C/D layout -> global (M = vgpr + (lane<16?0:8), N = lane&15) --------
    const int mBase = 16 * rb + ((lane < 16) ? 0 : 8);
    const int n0    = 32 * cp + l15;
#pragma unroll
    for (int v = 0; v < 8; ++v) {
        Ob[(size_t)(mBase + v) * 64u + n0]      = acc0[v] * scale;
        Ob[(size_t)(mBase + v) * 64u + n0 + 16] = acc1[v] * scale;
    }
}

// Final scalar stage: D, Sn, log-sum-exp margin loss. dot is 64x64, diag = sq.
__global__ __launch_bounds__(64)
void contrastive_loss_kernel(const float* __restrict__ dot,
                             const long long* __restrict__ target,
                             float* __restrict__ out) {
    __shared__ float sq[64];
    __shared__ float Sn[64];
    __shared__ int   tg[64];
    __shared__ float redS[64];
    __shared__ float redP[64];

    const int i = threadIdx.x;
    tg[i] = (int)target[i];
    sq[i] = dot[i * 64 + i];
    __syncthreads();

    const int   ti  = tg[i];
    const float sqi = sq[i];
    float sn = 0.f;
    for (int k = 0; k < 64; ++k) {
        if (tg[k] != ti) {
            float D = (sqi + sq[k] - 2.f * dot[i * 64 + k]) * (1.f / 4096.f);
            sn += __expf(ALPHA_C - D);
        }
    }
    Sn[i] = sn;
    __syncthreads();

    float ls = 0.f, pc = 0.f;
    for (int j = i + 1; j < 64; ++j) {
        if (tg[j] == ti) {
            float D = (sqi + sq[j] - 2.f * dot[i * 64 + j]) * (1.f / 4096.f);
            float J = __logf(Sn[i] + Sn[j]) + D;
            float r = fmaxf(J, 0.f);
            ls += r * r;
            pc += 1.f;
        }
    }
    redS[i] = ls;
    redP[i] = pc;
    __syncthreads();
    if (i == 0) {
        float s = 0.f, p = 0.f;
        for (int k = 0; k < 64; ++k) { s += redS[k]; p += redP[k]; }
        out[0] = (p > 0.f) ? s / (2.f * p) : 0.f;
    }
}

extern "C" void kernel_launch(void* const* d_in, const int* in_sizes, int n_in,
                              void* d_out, int out_size, void* d_ws, size_t ws_size,
                              hipStream_t stream) {
    const float*     inp = (const float*)d_in[0];       // [64,64,128,128] fp32
    const long long* tgt = (const long long*)d_in[1];   // [64] int64
    float* ws  = (float*)d_ws;
    float* G   = ws;                // 64 * 4096 floats (batched Grams, flattened)
    float* dot = ws + 64 * 4096;    // 64 * 64 floats  (pairwise G dot products)
    float* out = (float*)d_out;     // scalar loss

    // Stage 1: G[b] = x[b] x[b]^T / (C*W);  64 blocks, K = 128*128 = 16384
    gram64_bf16x2_kernel<<<64, 256, 0, stream>>>(inp, G, 16384,
                                                 1.0f / (64.0f * 16384.0f));
    // Stage 2: dot = G_flat G_flat^T;  1 block, K = 4096
    gram64_bf16x2_kernel<<<1, 256, 0, stream>>>(G, dot, 4096, 1.0f);
    // Stage 3: loss
    contrastive_loss_kernel<<<1, 64, 0, stream>>>(dot, tgt, out);
}
